// Neuralsymbol_46325517255369
// MI455X (gfx1250) — matrix operand order repaired and checked by reading, not compiled, not run on hardware
//
#include <hip/hip_runtime.h>
#include <hip/hip_bf16.h>

// ---------------- CDNA5 WMMA types ----------------
typedef __attribute__((ext_vector_type(16))) _Float16 v16h;
typedef __attribute__((ext_vector_type(8)))  _Float16 v8h;
typedef __attribute__((ext_vector_type(8)))  float    v8f;

#define WMMA_F16(a, b, c) \
  __builtin_amdgcn_wmma_f32_16x16x32_f16(false, (a), false, (b), (short)0, (c), false, false)

#define SEQ   2048
#define VDIM  128
#define HDIM  64
#define EPSV  1e-6f

// d_ws layout (f16 elements):
//   [0      , 147456) W1cat_t : [384 out][384 k]   (cols 0..255 = op heads 0..3, 256..383 = val heads 0..1)
//   [147456 , 163840) val_w2_t: [2][128 out][64 k]
//   [163840 , 196608) out_w1_t: [64 out][512 k]
//   [196608 , 204800) out_w2_t: [128 out][64 k]
#define OFF_W1    0
#define OFF_VW2   147456
#define OFF_OW1   163840
#define OFF_OW2   196608
#define N_PREP    204800

// ---------------- weight transpose/convert to f16 (once per launch) ----------------
__global__ void ns_prep(const float* __restrict__ op_w1, const float* __restrict__ val_w1,
                        const float* __restrict__ val_w2, const float* __restrict__ out_w1,
                        const float* __restrict__ out_w2, _Float16* __restrict__ ws) {
  int i = blockIdx.x * blockDim.x + threadIdx.x;
  if (i >= N_PREP) return;
  float v;
  if (i < OFF_VW2) {                       // W1cat_t[c][k]
    int c = i / 384, k = i % 384;
    if (c < 256) v = op_w1[((c >> 6) * 384 + k) * 64 + (c & 63)];
    else { int c2 = c - 256; v = val_w1[((c2 >> 6) * 384 + k) * 64 + (c2 & 63)]; }
  } else if (i < OFF_OW1) {                // val_w2_t[h][n][k]
    int j = i - OFF_VW2; int h = j >> 13; int r = j & 8191; int n = r >> 6; int k = r & 63;
    v = val_w2[(h * 64 + k) * 128 + n];
  } else if (i < OFF_OW2) {                // out_w1_t[n][k]
    int j = i - OFF_OW1; int n = j >> 9; int k = j & 511;
    v = out_w1[k * 64 + n];
  } else {                                 // out_w2_t[n][k]
    int j = i - OFF_OW2; int n = j >> 6; int k = j & 63;
    v = out_w2[k * 128 + n];
  }
  ws[i] = (_Float16)v;
}

// ---------------- fragment loaders (contiguous k order, same for A and B) ----------------
// A 16x32 f16: lane L -> row (L&15); K=(L>>4)*16..+15 contiguous
__device__ __forceinline__ v16h load_a_frag(const _Float16* base, int ld, int lane) {
  int m = lane & 15, half = lane >> 4;
  const _Float16* p = base + m * ld + half * 16;
  union { v16h v; v8h h[2]; } u;
  u.h[0] = *(const v8h*)(p);
  u.h[1] = *(const v8h*)(p + 8);
  return u.v;
}
// B 32x16 f16 stored transposed [N][K]: lane L -> col (L&15), K=(L>>4)*16..+15 contiguous
__device__ __forceinline__ v16h load_b_frag(const _Float16* wt, int ldk, int lane) {
  int n = lane & 15, half = lane >> 4;
  return *(const v16h*)(wt + n * ldk + half * 16);
}

// ---------------- main recurrent kernel: one block = 16 batch rows ----------------
__global__ __launch_bounds__(256) void ns_main(
    const float* __restrict__ x,
    const float* __restrict__ op_b1, const float* __restrict__ op_w2, const float* __restrict__ op_b2,
    const float* __restrict__ val_b1, const float* __restrict__ val_b2,
    const float* __restrict__ out_b1, const float* __restrict__ out_b2,
    const float* __restrict__ c_sharp, const float* __restrict__ w_sharp,
    const float* __restrict__ init_vec,
    const _Float16* __restrict__ ws, float* __restrict__ out) {

  __shared__ float    s_cstack[16][4][VDIM];
  __shared__ float    s_wstack[16][4][VDIM];
  __shared__ float    s_cptr[16][4];
  __shared__ float    s_wptr[16][4];
  __shared__ _Float16 s_hbuf[16][648];   // 0..127 c_peek | 128..255 w_peek | 256..383 x | 384..511 cval | 512..639 wval
  __shared__ _Float16 s_h1[16][392];     // fused layer-1 outputs (relu, f16)
  __shared__ float    s_val[2][16][VDIM];
  __shared__ _Float16 s_o1[16][72];
  __shared__ float    s_p[16][8];        // sigmoid head outputs

  const _Float16* W1cat_t  = ws + OFF_W1;
  const _Float16* val_w2_t = ws + OFF_VW2;
  const _Float16* out_w1_t = ws + OFF_OW1;
  const _Float16* out_w2_t = ws + OFF_OW2;

  const int b0   = blockIdx.x * 16;
  const int t    = threadIdx.x;
  const int wave = t >> 5, lane = t & 31;
  const int bt   = t >> 4;            // batch row for elementwise phases
  const int vb   = (t & 15) * 8;      // 8 V-elements per thread

  // ---- init: stacks = push(eps-stack, ptr0, iv) -> slot1 = normalized init_vec ----
  {
    float n2 = 0.f;
#pragma unroll 4
    for (int v = 0; v < VDIM; ++v) { float q = init_vec[v]; n2 += q * q; }
    float inv = rsqrtf(n2);
#pragma unroll
    for (int n = 0; n < 4; ++n)
#pragma unroll
      for (int j = 0; j < 8; ++j) {
        int v = vb + j;
        float sv = (n == 1) ? init_vec[v] * inv : EPSV;
        s_cstack[bt][n][v] = sv; s_wstack[bt][n][v] = sv;
      }
    if ((t & 15) == 0)
#pragma unroll
      for (int n = 0; n < 4; ++n) { s_cptr[bt][n] = (n == 1) ? 1.f : 0.f; s_wptr[bt][n] = (n == 1) ? 1.f : 0.f; }
  }
  __syncthreads();

  const float cs = c_sharp[0], wsp = w_sharp[0];

  for (int s = 0; s < SEQ; ++s) {
    // ---- A: peeks + stage x (f16 into LDS) ----
#pragma unroll
    for (int j = 0; j < 8; ++j) {
      int v = vb + j;
      float cp = 0.f, wp = 0.f;
#pragma unroll
      for (int n = 0; n < 4; ++n) {
        cp += s_cstack[bt][n][v] * s_cptr[bt][n];
        wp += s_wstack[bt][n][v] * s_wptr[bt][n];
      }
      s_hbuf[bt][v]       = (_Float16)cp;
      s_hbuf[bt][128 + v] = (_Float16)wp;
      s_hbuf[bt][256 + v] = (_Float16)x[((size_t)(b0 + bt) * SEQ + s) * VDIM + v];
    }
    __syncthreads();

    // ---- B: fused layer-1 GEMM  H1 = relu(h3 @ W1cat + b)  (16x384 @ 384x384) ----
#pragma unroll
    for (int ti = 0; ti < 3; ++ti) {
      int nt = wave * 3 + ti;
      v8f acc = {};
#pragma unroll
      for (int kk = 0; kk < 12; ++kk) {
        v16h a = load_a_frag(&s_hbuf[0][kk * 32], 648, lane);
        v16h b = load_b_frag(W1cat_t + (size_t)(nt * 16) * 384 + kk * 32, 384, lane);
        acc = WMMA_F16(a, b, acc);
      }
      int n = lane & 15, col = nt * 16 + n, mb = (lane >> 4) * 8;
      float bias = (col < 256) ? op_b1[col] : val_b1[col - 256];
#pragma unroll
      for (int r = 0; r < 8; ++r)
        s_h1[mb + r][col] = (_Float16)fmaxf(acc[r] + bias, 0.f);
    }
    __syncthreads();

    // ---- C: val heads GEMM (2x 16x64 @ 64x128, tanh) + op-head VALU dots (sigmoid) ----
#pragma unroll
    for (int ti = 0; ti < 2; ++ti) {
      int t2 = wave * 2 + ti, head = t2 >> 3, nt = t2 & 7;
      v8f acc = {};
#pragma unroll
      for (int kk = 0; kk < 2; ++kk) {
        v16h a = load_a_frag(&s_h1[0][256 + head * 64 + kk * 32], 392, lane);
        v16h b = load_b_frag(val_w2_t + head * 8192 + (nt * 16) * 64 + kk * 32, 64, lane);
        acc = WMMA_F16(a, b, acc);
      }
      int n = lane & 15, col = nt * 16 + n, mb = (lane >> 4) * 8;
      float bias = val_b2[head * 128 + col];
#pragma unroll
      for (int r = 0; r < 8; ++r) {
        float vo = tanhf(acc[r] + bias);
        int m = mb + r;
        s_val[head][m][col] = vo;
        s_hbuf[m][384 + head * 128 + col] = (_Float16)vo;  // h4 region
      }
    }
    if (t < 128) {  // 16 rows x 4 heads x 2 outputs, dim-64 dots
      int b = t >> 3, idx = t & 7, head = idx >> 1, j = idx & 1;
      float z = op_b2[head * 2 + j];
#pragma unroll 8
      for (int i = 0; i < 64; ++i)
        z += (float)s_h1[b][head * 64 + i] * op_w2[(head * 64 + i) * 2 + j];
      s_p[b][idx] = 1.f / (1.f + expf(-z));
    }
    __syncthreads();

    // ---- D: snapshot ptrs + probs into registers ----
    float cp4[4], wp4[4];
#pragma unroll
    for (int n = 0; n < 4; ++n) { cp4[n] = s_cptr[bt][n]; wp4[n] = s_wptr[bt][n]; }
    float pcp = s_p[bt][0], pcn = s_p[bt][1], pwp = s_p[bt][2], pwn = s_p[bt][3];
    float qcp = s_p[bt][4], qcn = s_p[bt][5], qwp = s_p[bt][6], qwn = s_p[bt][7];
    __syncthreads();

    // ---- E: fused pop-mix + push-mix stack update (VALU)  ||  out-MLP layer1 GEMM ----
    {
      float paC1 = 1.f / (1.f + expf((pcn - pcp) * cs)), pbC1 = 1.f - paC1;
      float paW1 = 1.f / (1.f + expf((pwn - pwp) * wsp)), pbW1 = 1.f - paW1;
      float paC2 = 1.f / (1.f + expf((qcn - qcp) * cs)), pbC2 = 1.f - paC2;
      float paW2 = 1.f / (1.f + expf((qwn - qwp) * wsp)), pbW2 = 1.f - paW2;
      float c1p[4], w1p[4], c2r[4], w2r[4], cnp[4], wnp[4];
#pragma unroll
      for (int n = 0; n < 4; ++n) {
        c1p[n] = paC1 * cp4[(n + 1) & 3] + pbC1 * cp4[n];   // pop-mixed ptr
        w1p[n] = paW1 * wp4[(n + 1) & 3] + pbW1 * wp4[n];
      }
#pragma unroll
      for (int n = 0; n < 4; ++n) {
        c2r[n] = c1p[(n + 3) & 3];                          // push-rolled ptr
        w2r[n] = w1p[(n + 3) & 3];
        cnp[n] = paC2 * c2r[n] + pbC2 * c1p[n];
        wnp[n] = paW2 * w2r[n] + pbW2 * w1p[n];
      }
#pragma unroll
      for (int n = 0; n < 4; ++n)
#pragma unroll
        for (int j = 0; j < 8; ++j) {
          int v = vb + j;
          float sc = s_cstack[bt][n][v];
          float s1 = paC1 * (sc * (1.f - cp4[n]) + EPSV * cp4[n]) + pbC1 * sc;
          float s2 = paC2 * (s1 * (1.f - c2r[n]) + s_val[0][bt][v] * c2r[n]) + pbC2 * s1;
          s_cstack[bt][n][v] = s2;
          float sw = s_wstack[bt][n][v];
          float t1 = paW1 * (sw * (1.f - wp4[n]) + EPSV * wp4[n]) + pbW1 * sw;
          float t2 = paW2 * (t1 * (1.f - w2r[n]) + s_val[1][bt][v] * w2r[n]) + pbW2 * t1;
          s_wstack[bt][n][v] = t2;
        }
      if ((t & 15) == 0)
#pragma unroll
        for (int n = 0; n < 4; ++n) { s_cptr[bt][n] = cnp[n]; s_wptr[bt][n] = wnp[n]; }
    }
    if (wave < 4) {  // o1 = relu(h4 @ out_w1 + b)  (16x512 @ 512x64)
      int nt = wave;
      v8f acc = {};
#pragma unroll
      for (int kk = 0; kk < 16; ++kk) {
        int cb = (kk < 8) ? kk * 32 : kk * 32 + 128;       // skip x-region of hbuf
        v16h a = load_a_frag(&s_hbuf[0][cb], 648, lane);
        v16h b = load_b_frag(out_w1_t + (nt * 16) * 512 + kk * 32, 512, lane);
        acc = WMMA_F16(a, b, acc);
      }
      int n = lane & 15, col = nt * 16 + n, mb = (lane >> 4) * 8;
      float bias = out_b1[col];
#pragma unroll
      for (int r = 0; r < 8; ++r)
        s_o1[mb + r][col] = (_Float16)fmaxf(acc[r] + bias, 0.f);
    }
    __syncthreads();

    // ---- F: out = tanh(o1 @ out_w2 + b)  (16x64 @ 64x128) -> global ----
    {
      int nt = wave;
      v8f acc = {};
#pragma unroll
      for (int kk = 0; kk < 2; ++kk) {
        v16h a = load_a_frag(&s_o1[0][kk * 32], 72, lane);
        v16h b = load_b_frag(out_w2_t + (nt * 16) * 64 + kk * 32, 64, lane);
        acc = WMMA_F16(a, b, acc);
      }
      int n = lane & 15, col = nt * 16 + n, mb = (lane >> 4) * 8;
      float bias = out_b2[col];
#pragma unroll
      for (int r = 0; r < 8; ++r) {
        int m = mb + r;
        out[((size_t)(b0 + m) * SEQ + s) * VDIM + col] = tanhf(acc[r] + bias);
      }
    }
    // next iteration's A/B/C/D barriers protect s_o1 / s_hbuf reuse
  }
}

extern "C" void kernel_launch(void* const* d_in, const int* in_sizes, int n_in,
                              void* d_out, int out_size, void* d_ws, size_t ws_size,
                              hipStream_t stream) {
  const float* x        = (const float*)d_in[0];
  const float* op_w1    = (const float*)d_in[1];
  const float* op_b1    = (const float*)d_in[2];
  const float* op_w2    = (const float*)d_in[3];
  const float* op_b2    = (const float*)d_in[4];
  const float* val_w1   = (const float*)d_in[5];
  const float* val_b1   = (const float*)d_in[6];
  const float* val_w2   = (const float*)d_in[7];
  const float* val_b2   = (const float*)d_in[8];
  const float* out_w1   = (const float*)d_in[9];
  const float* out_b1   = (const float*)d_in[10];
  const float* out_w2   = (const float*)d_in[11];
  const float* out_b2   = (const float*)d_in[12];
  const float* c_sharp  = (const float*)d_in[13];
  const float* w_sharp  = (const float*)d_in[14];
  const float* init_vec = (const float*)d_in[15];
  _Float16* ws = (_Float16*)d_ws;   // needs N_PREP*2 = 409,600 bytes

  ns_prep<<<(N_PREP + 255) / 256, 256, 0, stream>>>(op_w1, val_w1, val_w2, out_w1, out_w2, ws);
  ns_main<<<4, 256, 0, stream>>>(x, op_b1, op_w2, op_b2, val_b1, val_b2,
                                 out_b1, out_b2, c_sharp, w_sharp, init_vec,
                                 ws, (float*)d_out);
}
